// LlamaMoEForCausalLM_30425548325402
// MI455X (gfx1250) — compile-verified, matrix operand order checked by reading, not run on hardware
//
#include <hip/hip_runtime.h>
#include <math.h>

// ---------------- problem constants (match reference) ----------------
constexpr int HDIM = 2048;   // hidden
constexpr int IDIM = 5632;   // intermediate
constexpr int NEXP = 8;      // experts
constexpr int TTOK = 2048;   // tokens (2*1024)
constexpr int TOPK = 2;

// ---------------- tiling ----------------
constexpr int BM  = 32;   // rows per block
constexpr int BN1 = 64;   // cols per block (gemm1, per g/u half)
constexpr int BN2 = 128;  // cols per block (gemm2)
constexpr int BK  = 64;   // k per LDS stage (2 bf16-WMMA K-steps)

typedef __bf16 bf16_t;
typedef bf16_t v2bf  __attribute__((ext_vector_type(2)));
typedef bf16_t v8bf  __attribute__((ext_vector_type(8)));
typedef bf16_t v16bf __attribute__((ext_vector_type(16)));
typedef float  v8f   __attribute__((ext_vector_type(8)));

// packed fp32x2 -> bf16x2 (native v_cvt_pk_bf16_f32), returned as one dword
__device__ __forceinline__ unsigned pkbf(float a, float b) {
  v2bf p;
  p.x = (bf16_t)a;
  p.y = (bf16_t)b;
  unsigned r;
  __builtin_memcpy(&r, &p, 4);
  return r;
}

__device__ __forceinline__ v16bf cat16(v8bf lo, v8bf hi) {
  return __builtin_shufflevector(lo, hi, 0,1,2,3,4,5,6,7,8,9,10,11,12,13,14,15);
}

__device__ __forceinline__ float fast_silu(float g) {
  // g * sigmoid(g) with v_rcp_f32 instead of an IEEE divide ladder
  return g * __builtin_amdgcn_rcpf(1.0f + __expf(-g));
}

// =====================================================================
// 0) zero output + atomic counters (out/ws are poisoned by harness)
// =====================================================================
__global__ __launch_bounds__(256) void moe_zero(float* __restrict__ out, int n,
                                                int* __restrict__ meta) {
  int i = blockIdx.x * 256 + threadIdx.x;
  if (i < n) out[i] = 0.0f;
  if (i < 32) meta[i] = 0;   // counts[8], cursors[8], offsets[9]
}

// =====================================================================
// 1) router: one wave per token. logits = x @ gate_w^T, softmax, top-2
// =====================================================================
__global__ __launch_bounds__(256) void moe_router(const float* __restrict__ x,
                                                  const float* __restrict__ gate_w,
                                                  int* __restrict__ counts,
                                                  int* __restrict__ tok_e,
                                                  float* __restrict__ tok_w) {
  const int wave = threadIdx.x >> 5;
  const int lane = threadIdx.x & 31;
  const int t = blockIdx.x * 8 + wave;
  if (t >= TTOK) return;

  float acc[NEXP];
#pragma unroll
  for (int e = 0; e < NEXP; ++e) acc[e] = 0.0f;

  for (int h = lane; h < HDIM; h += 32) {
    float xv = x[(size_t)t * HDIM + h];
#pragma unroll
    for (int e = 0; e < NEXP; ++e) acc[e] += xv * gate_w[e * HDIM + h];
  }
#pragma unroll
  for (int e = 0; e < NEXP; ++e) {
    float v = acc[e];
#pragma unroll
    for (int off = 16; off > 0; off >>= 1) v += __shfl_xor(v, off, 32);
    acc[e] = v;
  }
  if (lane == 0) {
    float mx = acc[0];
#pragma unroll
    for (int e = 1; e < NEXP; ++e) mx = fmaxf(mx, acc[e]);
    float p[NEXP];
#pragma unroll
    for (int e = 0; e < NEXP; ++e) p[e] = __expf(acc[e] - mx);
    // top-2 (renormalization over top-2 cancels the softmax denominator)
    int e0 = 0; float b0 = p[0];
#pragma unroll
    for (int e = 1; e < NEXP; ++e) if (p[e] > b0) { b0 = p[e]; e0 = e; }
    int e1 = (e0 == 0) ? 1 : 0; float b1 = p[e1];
#pragma unroll
    for (int e = 0; e < NEXP; ++e)
      if (e != e0 && p[e] > b1) { b1 = p[e]; e1 = e; }
    float inv = __builtin_amdgcn_rcpf(b0 + b1);
    tok_e[2 * t]     = e0;  tok_w[2 * t]     = b0 * inv;
    tok_e[2 * t + 1] = e1;  tok_w[2 * t + 1] = b1 * inv;
    atomicAdd(&counts[e0], 1);
    atomicAdd(&counts[e1], 1);
  }
}

// =====================================================================
// 2) exclusive scan of 8 counts (tiny, one thread)
// =====================================================================
__global__ void moe_scan(const int* __restrict__ counts, int* __restrict__ offsets) {
  if (threadIdx.x == 0 && blockIdx.x == 0) {
    int a = 0;
    for (int e = 0; e < NEXP; ++e) { offsets[e] = a; a += counts[e]; }
    offsets[NEXP] = a;
  }
}

// =====================================================================
// 3) fill per-expert slot lists
// =====================================================================
__global__ __launch_bounds__(256) void moe_fill(const int* __restrict__ tok_e,
                                                const float* __restrict__ tok_w,
                                                const int* __restrict__ offsets,
                                                int* __restrict__ cursors,
                                                int* __restrict__ slot_token,
                                                float* __restrict__ slot_weight) {
  int t = blockIdx.x * 256 + threadIdx.x;
  if (t >= TTOK) return;
#pragma unroll
  for (int k = 0; k < TOPK; ++k) {
    int e = tok_e[2 * t + k];
    int pos = atomicAdd(&cursors[e], 1);
    int s = offsets[e] + pos;
    slot_token[s]  = t;
    slot_weight[s] = tok_w[2 * t + k];
  }
}

// =====================================================================
// 4) GEMM1: gu = gather(x) @ W_gate_up ; act = silu(g)*u  (bf16 out)
//    segment z: 0..7 experts, 8 = dense base MLP
//    block tile 32x64 (g) + 32x64 (u), BK=64 => 4 WMMA / wave / stage
// =====================================================================
__global__ __launch_bounds__(256) void moe_gemm1(const float* __restrict__ x,
                                                 const float* __restrict__ base_gu,
                                                 const float* __restrict__ exp_gu,
                                                 const int* __restrict__ counts,
                                                 const int* __restrict__ offsets,
                                                 const int* __restrict__ slot_token,
                                                 bf16_t* __restrict__ act) {
  __shared__ __align__(32) bf16_t As[BM][BK];    // 4 KB
  __shared__ __align__(32) bf16_t Bg[BN1][BK];   // 8 KB, transposed [n][k]
  __shared__ __align__(32) bf16_t Bu[BN1][BK];   // 8 KB
  __shared__ int rowtok[BM];

  const int seg = blockIdx.z;
  int cnt, rowStart;
  const float* W;
  if (seg == NEXP) { cnt = TTOK; rowStart = 0; W = base_gu; }
  else {
    cnt = counts[seg];
    rowStart = offsets[seg];
    W = exp_gu + (size_t)seg * HDIM * (2 * IDIM);
  }
  const int m0 = blockIdx.y * BM;
  if (m0 >= cnt) return;
  const int n0 = blockIdx.x * BN1;
  const int actBase = (seg == NEXP) ? 0 : (TTOK + rowStart);

  const int tid = threadIdx.x;
  if (tid < BM) {
    int m = m0 + tid;
    int t;
    if (seg == NEXP) t = (m < cnt) ? m : 0;
    else             t = (m < cnt) ? slot_token[rowStart + m] : 0;
    rowtok[tid] = t;
  }
  __syncthreads();

  const int wave = tid >> 5, lane = tid & 31;
  const int wm = wave & 1, wn = wave >> 1;        // 2x4 wave grid -> 32x64
  const int half = lane >> 4;                     // K half per ISA layout
  const int mrow = wm * 16 + (lane & 15);
  const int bcol = wn * 16 + (lane & 15);

  v8f accg = {}; v8f accu = {};

  for (int k0 = 0; k0 < HDIM; k0 += BK) {
    // ---- stage A: 32x64 fp32 -> packed bf16x2 dword stores ----
#pragma unroll
    for (int j = 0; j < 4; ++j) {
      int p = tid + j * 256;              // pair index, 1024 pairs
      int r = p >> 5, kp = p & 31, k = kp * 2;
      const float* xr = &x[(size_t)rowtok[r] * HDIM + k0 + k];
      *(unsigned*)&As[r][k] = pkbf(xr[0], xr[1]);
    }
    // ---- stage B gate & up tiles (64k x 64n each), transposed ----
#pragma unroll
    for (int j = 0; j < 8; ++j) {
      int p = tid + j * 256;              // 2048 pairs per tile
      int nn = p & 63, kp = p >> 6, k = kp * 2;
      const float* w0 = W + (size_t)(k0 + k) * (2 * IDIM) + n0;
      const float* w1 = w0 + 2 * IDIM;
      if (j == 0) __builtin_prefetch(w0 + (size_t)BK * 2 * IDIM, 0, 1);
      *(unsigned*)&Bg[nn][k] = pkbf(w0[nn], w1[nn]);
      *(unsigned*)&Bu[nn][k] = pkbf(w0[IDIM + nn], w1[IDIM + nn]);
    }
    __syncthreads();

#pragma unroll
    for (int ks = 0; ks < BK; ks += 32) {
      v16bf a  = cat16(*(const v8bf*)&As[mrow][ks + half * 8],
                       *(const v8bf*)&As[mrow][ks + 16 + half * 8]);
      v16bf bg = cat16(*(const v8bf*)&Bg[bcol][ks + half * 8],
                       *(const v8bf*)&Bg[bcol][ks + 16 + half * 8]);
      v16bf bu = cat16(*(const v8bf*)&Bu[bcol][ks + half * 8],
                       *(const v8bf*)&Bu[bcol][ks + 16 + half * 8]);
      accg = __builtin_amdgcn_wmma_f32_16x16x32_bf16(false, a, false, bg,
                                                     (short)0, accg, false, false);
      accu = __builtin_amdgcn_wmma_f32_16x16x32_bf16(false, a, false, bu,
                                                     (short)0, accu, false, false);
    }
    __syncthreads();
  }

  // ---- epilogue: act = silu(g)*u, native bf16 store ----
  const int lrow0 = wm * 16 + half * 8;
  const int lcol  = wn * 16 + (lane & 15);
#pragma unroll
  for (int r = 0; r < 8; ++r) {
    int mloc = lrow0 + r;
    if (m0 + mloc < cnt) {
      float v = fast_silu(accg[r]) * accu[r];
      act[(size_t)(actBase + m0 + mloc) * IDIM + n0 + lcol] = (bf16_t)v;
    }
  }
}

// =====================================================================
// 5) GEMM2: out += weight * (act @ W_down)   (atomic fp32 combine)
//    block tile 32x128, BK=64, 2 N-subtiles/wave => 4 WMMA / wave / stage
//    A tile staged with GLOBAL_LOAD_ASYNC_TO_LDS_B128 (bf16 pass-through),
//    overlapped with the fp32->bf16 conversion of the B tile.
// =====================================================================
__global__ __launch_bounds__(256) void moe_gemm2(const bf16_t* __restrict__ act,
                                                 const float* __restrict__ base_down,
                                                 const float* __restrict__ exp_down,
                                                 const int* __restrict__ counts,
                                                 const int* __restrict__ offsets,
                                                 const int* __restrict__ slot_token,
                                                 const float* __restrict__ slot_weight,
                                                 float* __restrict__ out) {
  __shared__ __align__(32) bf16_t As[BM][BK];    // 4 KB
  __shared__ __align__(32) bf16_t Bt[BN2][BK];   // 16 KB, transposed [n][k]
  __shared__ int   rowtok[BM];
  __shared__ float roww[BM];

  const int seg = blockIdx.z;
  int cnt, rowStart;
  const float* W;
  if (seg == NEXP) { cnt = TTOK; rowStart = 0; W = base_down; }
  else {
    cnt = counts[seg];
    rowStart = offsets[seg];
    W = exp_down + (size_t)seg * IDIM * HDIM;
  }
  const int m0 = blockIdx.y * BM;
  if (m0 >= cnt) return;
  const int n0 = blockIdx.x * BN2;

  const int tid = threadIdx.x;
  if (tid < BM) {
    int m = m0 + tid;
    bool v = m < cnt;
    if (seg == NEXP) { rowtok[tid] = v ? m : 0; roww[tid] = v ? 1.0f : 0.0f; }
    else {
      rowtok[tid] = v ? slot_token[rowStart + m] : 0;
      roww[tid]   = v ? slot_weight[rowStart + m] : 0.0f;
    }
  }
  __syncthreads();

  const int wave = tid >> 5, lane = tid & 31;
  const int wm = wave & 1, wn = wave >> 1;       // 2 x 4 waves, 2 n-subtiles each
  const int half = lane >> 4;
  const int mrow  = wm * 16 + (lane & 15);
  const int bcol0 = wn * 32 + (lane & 15);
  const int bcol1 = bcol0 + 16;

  // per-thread A-tile async copy coordinates (one b128 per thread)
  const int a_r = tid >> 3, a_koff = (tid & 7) * 8;   // 32 rows x 8 chunks
  int a_rr = m0 + a_r; if (a_rr >= cnt) a_rr = cnt - 1;
  const size_t a_row = (seg == NEXP) ? (size_t)a_rr
                                     : (size_t)(TTOK + rowStart + a_rr);
  const unsigned a_lds = (unsigned)(uintptr_t)&As[a_r][a_koff];

  v8f acc0 = {}; v8f acc1 = {};

  for (int k0 = 0; k0 < IDIM; k0 += BK) {
    // ---- stage A: async DMA global(bf16) -> LDS, no VGPR round-trip ----
    {
      const bf16_t* gsrc = &act[a_row * IDIM + k0 + a_koff];
      asm volatile("global_load_async_to_lds_b128 %0, %1, off"
                   :: "v"(a_lds), "v"(gsrc) : "memory");
    }
    // ---- stage B: 64k x 128n fp32 -> packed bf16, transposed (overlaps DMA) ----
#pragma unroll
    for (int j = 0; j < 16; ++j) {
      int p = tid + j * 256;                     // 4096 pairs
      int nn = p & 127, kp = p >> 7, k = kp * 2;
      const float* w0 = W + (size_t)(k0 + k) * HDIM + n0;
      if (j == 0) __builtin_prefetch(w0 + (size_t)BK * HDIM, 0, 1);
      *(unsigned*)&Bt[nn][k] = pkbf(w0[nn], w0[HDIM + nn]);
    }
    asm volatile("s_wait_asynccnt 0" ::: "memory");
    __syncthreads();

#pragma unroll
    for (int ks = 0; ks < BK; ks += 32) {
      v16bf a  = cat16(*(const v8bf*)&As[mrow][ks + half * 8],
                       *(const v8bf*)&As[mrow][ks + 16 + half * 8]);
      v16bf b0 = cat16(*(const v8bf*)&Bt[bcol0][ks + half * 8],
                       *(const v8bf*)&Bt[bcol0][ks + 16 + half * 8]);
      v16bf b1 = cat16(*(const v8bf*)&Bt[bcol1][ks + half * 8],
                       *(const v8bf*)&Bt[bcol1][ks + 16 + half * 8]);
      acc0 = __builtin_amdgcn_wmma_f32_16x16x32_bf16(false, a, false, b0,
                                                     (short)0, acc0, false, false);
      acc1 = __builtin_amdgcn_wmma_f32_16x16x32_bf16(false, a, false, b1,
                                                     (short)0, acc1, false, false);
    }
    __syncthreads();
  }

  const int lrow0 = wm * 16 + half * 8;
  const int lcol0 = wn * 32 + (lane & 15);
#pragma unroll
  for (int r = 0; r < 8; ++r) {
    int mloc = lrow0 + r;
    if (m0 + mloc < cnt) {
      float w = roww[mloc];
      int tok = rowtok[mloc];
      atomicAdd(&out[(size_t)tok * HDIM + n0 + lcol0],      acc0[r] * w);
      atomicAdd(&out[(size_t)tok * HDIM + n0 + lcol0 + 16], acc1[r] * w);
    }
  }
}

// =====================================================================
// launch
// =====================================================================
extern "C" void kernel_launch(void* const* d_in, const int* in_sizes, int n_in,
                              void* d_out, int out_size, void* d_ws, size_t ws_size,
                              hipStream_t stream) {
  const float* x              = (const float*)d_in[0];
  const float* gate_w         = (const float*)d_in[1];
  const float* base_gate_up   = (const float*)d_in[2];
  const float* base_down      = (const float*)d_in[3];
  const float* expert_gate_up = (const float*)d_in[4];
  const float* expert_down    = (const float*)d_in[5];
  float* out = (float*)d_out;

  // ---- workspace layout ----
  char* ws = (char*)d_ws;
  int*   meta        = (int*)ws;                 // counts[8] | cursors[8] | offsets[9]
  int*   counts      = meta;
  int*   cursors     = meta + 8;
  int*   offsets     = meta + 16;
  const size_t SLOTS = (size_t)TOPK * TTOK;      // 4096 routed slots
  int*   tok_e       = (int*)  (ws + 1024);
  float* tok_w       = (float*)(ws + 1024 + 1 * SLOTS * 4);
  int*   slot_token  = (int*)  (ws + 1024 + 2 * SLOTS * 4);
  float* slot_weight = (float*)(ws + 1024 + 3 * SLOTS * 4);
  bf16_t* act        = (bf16_t*)(ws + 1024 + 4 * SLOTS * 4);  // (T + 2T) x I bf16

  (void)in_sizes; (void)n_in; (void)out_size; (void)ws_size;

  moe_zero<<<(TTOK * HDIM + 255) / 256, 256, 0, stream>>>(out, TTOK * HDIM, meta);
  moe_router<<<TTOK / 8, 256, 0, stream>>>(x, gate_w, counts, tok_e, tok_w);
  moe_scan<<<1, 32, 0, stream>>>(counts, offsets);
  moe_fill<<<(TTOK + 255) / 256, 256, 0, stream>>>(tok_e, tok_w, offsets, cursors,
                                                   slot_token, slot_weight);
  moe_gemm1<<<dim3(IDIM / BN1, TTOK / BM, NEXP + 1), 256, 0, stream>>>(
      x, base_gate_up, expert_gate_up, counts, offsets, slot_token, act);
  moe_gemm2<<<dim3(HDIM / BN2, TTOK / BM, NEXP + 1), 256, 0, stream>>>(
      act, base_down, expert_down, counts, offsets, slot_token, slot_weight, out);
}
